// ChatDecoder_74912819577096
// MI455X (gfx1250) — matrix operand order repaired and checked by reading, not compiled, not run on hardware
//
#include <hip/hip_runtime.h>

// Problem constants (from the reference)
#define B_  64      // batch
#define U_  512     // latent units (= E)
#define E_  512     // embedding dim
#define V_  32000   // vocab
#define G_  2048    // 4*U gates
#define T_  20      // decode steps

typedef __bf16 bf16x16 __attribute__((ext_vector_type(16)));
typedef float  f32x8   __attribute__((ext_vector_type(8)));
typedef int    v4i_    __attribute__((vector_size(16)));

union BF16Vec {
  bf16x16 v;
  unsigned int u[8];
};

#if defined(__has_builtin)
#  if __has_builtin(__builtin_amdgcn_global_load_async_to_lds_b128)
#    define USE_ASYNC_LDS 1
#  endif
#  if __has_builtin(__builtin_amdgcn_s_wait_asynccnt)
#    define HAVE_WAIT_ASYNC 1
#  endif
#endif

__device__ __forceinline__ unsigned short f32_to_bf16_rne(float f) {
  unsigned int u = __float_as_uint(f);
  u += 0x7FFFu + ((u >> 16) & 1u);      // round-to-nearest-even
  return (unsigned short)(u >> 16);
}
__device__ __forceinline__ f32x8 zero8() {
  f32x8 z;
#pragma unroll
  for (int i = 0; i < 8; ++i) z[i] = 0.0f;
  return z;
}
__device__ __forceinline__ float sigmoidf_(float x) {
  return 1.0f / (1.0f + __expf(-x));
}

// ---------------------------------------------------------------------------
// src is [512, N] row-major fp32; dst is [N, 512] row-major bf16 (K-major per
// output column). Consecutive tids write consecutive dst elements.
// ---------------------------------------------------------------------------
__global__ void k_convert_transpose(const float* __restrict__ src,
                                    unsigned short* __restrict__ dst,
                                    int N, int total) {
  int tid = blockIdx.x * blockDim.x + threadIdx.x;
  if (tid >= total) return;
  int n = tid >> 9;       // / 512
  int k = tid & 511;
  dst[tid] = f32_to_bf16_rne(src[(size_t)k * N + n]);
}

__global__ void k_init_state(const float* __restrict__ h0,
                             const float* __restrict__ c0,
                             unsigned short* __restrict__ hbf,
                             float* __restrict__ c, int* __restrict__ idx) {
  int tid = blockIdx.x * blockDim.x + threadIdx.x;   // B_*U_ threads
  c[tid] = c0[tid];
  hbf[tid] = f32_to_bf16_rne(h0[tid]);
  if (tid < B_) idx[tid] = 1;                        // GO token
}

// ---------------------------------------------------------------------------
// Embedding gather + fp32->bf16 convert, hoisted out of the GEMM K-loop so
// the GEMM inner loops are pure b128-load + v_wmma.
// ---------------------------------------------------------------------------
__global__ void k_embed(const float* __restrict__ emb,
                        const int* __restrict__ idx,
                        unsigned short* __restrict__ xbf) {
  int tid = blockIdx.x * blockDim.x + threadIdx.x;   // B_*E_ threads
  int b = tid >> 9;
  int k = tid & 511;
  xbf[tid] = f32_to_bf16_rne(emb[(size_t)idx[b] * E_ + k]);
}

// ---------------------------------------------------------------------------
// z = x_bf16 @ WxT + h_bf16 @ WhT + b   -> [64, 2048] fp32
// 512 waves; wave w owns C tile (mt = w>>7, nt = w&127).
// ---------------------------------------------------------------------------
__global__ void __launch_bounds__(256)
k_gates_gemm(const unsigned short* __restrict__ xbf,
             const unsigned short* __restrict__ WxT,
             const unsigned short* __restrict__ WhT,
             const unsigned short* __restrict__ hbf,
             const float* __restrict__ b_lstm,
             float* __restrict__ z) {
  const int lane   = threadIdx.x & 31;
  const int wave   = threadIdx.x >> 5;
  const int w      = blockIdx.x * 8 + wave;   // 0..511
  const int mt     = w >> 7;                  // 0..3
  const int nt     = w & 127;                 // 0..127
  const int lane15 = lane & 15;
  const int hiA    = (lane >> 4) * 8;         // A-fragment K sub-offset
  const int hiB    = (lane >> 4) * 16;        // unused; B fragments also use hiA
  const int rowm   = mt * 16 + lane15;        // batch row (A operand)
  const int n      = nt * 16 + lane15;        // output column (B/C/D)

  f32x8 acc = zero8();

  const unsigned short* xr  = xbf + (size_t)rowm * E_;
  const unsigned short* wxc = WxT + (size_t)n * E_;
  for (int kc = 0; kc < E_; kc += 32) {
    BF16Vec A, Bm;
    const unsigned short* ap = xr + kc + hiA;
#pragma unroll
    for (int v2 = 0; v2 < 4; ++v2) {
      A.u[v2]     = *(const unsigned int*)(ap + 2 * v2);
      A.u[v2 + 4] = *(const unsigned int*)(ap + 16 + 2 * v2);
    }
    const unsigned short* bp = wxc + kc + hiA;
#pragma unroll
    for (int v2 = 0; v2 < 4; ++v2) {
      Bm.u[v2]     = *(const unsigned int*)(bp + 2 * v2);
      Bm.u[v2 + 4] = *(const unsigned int*)(bp + 16 + 2 * v2);
    }
    acc = __builtin_amdgcn_wmma_f32_16x16x32_bf16(false, A.v, false, Bm.v,
                                                  (short)0, acc, false, false);
  }

  const unsigned short* hr  = hbf + (size_t)rowm * U_;
  const unsigned short* whc = WhT + (size_t)n * U_;
  for (int kc = 0; kc < U_; kc += 32) {
    BF16Vec A, Bm;
    const unsigned short* ap = hr + kc + hiA;
#pragma unroll
    for (int v2 = 0; v2 < 4; ++v2) {
      A.u[v2]     = *(const unsigned int*)(ap + 2 * v2);
      A.u[v2 + 4] = *(const unsigned int*)(ap + 16 + 2 * v2);
    }
    const unsigned short* bp = whc + kc + hiA;
#pragma unroll
    for (int v2 = 0; v2 < 4; ++v2) {
      Bm.u[v2]     = *(const unsigned int*)(bp + 2 * v2);
      Bm.u[v2 + 4] = *(const unsigned int*)(bp + 16 + 2 * v2);
    }
    acc = __builtin_amdgcn_wmma_f32_16x16x32_bf16(false, A.v, false, Bm.v,
                                                  (short)0, acc, false, false);
  }

  const float bias = b_lstm[n];
  const int rbase = mt * 16 + hiA;             // C/D row base for this lane
#pragma unroll
  for (int r = 0; r < 8; ++r)
    z[(size_t)(rbase + r) * G_ + n] = acc[r] + bias;
}

// ---------------------------------------------------------------------------
// LSTM cell update (gate order i, f, g, o), writes c (fp32) and h (bf16).
// ---------------------------------------------------------------------------
__global__ void k_lstm_update(const float* __restrict__ z,
                              float* __restrict__ c,
                              unsigned short* __restrict__ hbf) {
  int tid = blockIdx.x * blockDim.x + threadIdx.x;   // B_*U_
  int b = tid >> 9;
  int u = tid & 511;
  const float* zb = z + (size_t)b * G_;
  float zi = zb[u], zf = zb[512 + u], zg = zb[1024 + u], zo = zb[1536 + u];
  float cn = sigmoidf_(zf) * c[tid] + sigmoidf_(zi) * tanhf(zg);
  float hn = sigmoidf_(zo) * tanhf(cn);
  c[tid] = cn;
  hbf[tid] = f32_to_bf16_rne(hn);
}

// ---------------------------------------------------------------------------
// logits = h_bf16 @ WdT + bd -> fp32 straight into d_out[b, t, :].
// 500 blocks x 8 waves. A block owns 64 vocab columns for all 64 rows; the
// 64x512 bf16 Wd slab (64 KB) is staged once into LDS via async-to-LDS and
// then shared by all 8 waves -- the 4 M-tiles reuse each B fragment from LDS
// instead of 4x L2 traffic.
// ---------------------------------------------------------------------------
__global__ void __launch_bounds__(256)
k_logits_gemm(const unsigned short* __restrict__ hbf,
              const unsigned short* __restrict__ WdT,
              const float* __restrict__ bd,
              float* __restrict__ out, int t) {
  __shared__ unsigned short ldsB[64 * 512];          // 64 KB

  const int lane   = threadIdx.x & 31;
  const int wave   = threadIdx.x >> 5;
  const int mt     = wave >> 1;                      // 0..3
  const int lane15 = lane & 15;
  const int hiA    = (lane >> 4) * 8;
  const int hiB    = (lane >> 4) * 16;
  const int colLA  = (wave & 1) * 32 + lane15;       // block-local column
  const int nA     = blockIdx.x * 64 + colLA;
  const int nB     = nA + 16;
  const int rowm   = mt * 16 + lane15;

  // ---- Stage Wd columns [blockIdx*64, +64) x K=512 into LDS ----
  const unsigned short* gB = WdT + (size_t)blockIdx.x * 64 * U_;
#if defined(USE_ASYNC_LDS)
#pragma unroll
  for (int i = 0; i < 16; ++i) {
    int cid = threadIdx.x + i * 256;                 // 4096 x 16B chunks
    __builtin_amdgcn_global_load_async_to_lds_b128(
        (__attribute__((address_space(1))) v4i_*)(void*)(gB + cid * 8),
        (__attribute__((address_space(3))) v4i_*)(void*)(ldsB + cid * 8),
        0, 0);
  }
#  if defined(HAVE_WAIT_ASYNC)
  __builtin_amdgcn_s_wait_asynccnt(0);
#  else
  asm volatile("s_wait_asynccnt 0x0" ::: "memory");
#  endif
#else
#pragma unroll
  for (int i = 0; i < 16; ++i) {
    int cid = threadIdx.x + i * 256;
    *(uint4*)(ldsB + cid * 8) = *(const uint4*)(gB + cid * 8);
  }
#endif
  __syncthreads();

  f32x8 acc0 = zero8(), acc1 = zero8();
  const unsigned short* hr = hbf + (size_t)rowm * U_;
  const unsigned short* lA = ldsB + (size_t)colLA * U_;
  const unsigned short* lB = lA + 16 * U_;

  for (int kc = 0; kc < U_; kc += 32) {
    BF16Vec A, B0, B1;
    const unsigned short* ap = hr + kc + hiA;
#pragma unroll
    for (int v2 = 0; v2 < 4; ++v2) {
      A.u[v2]     = *(const unsigned int*)(ap + 2 * v2);
      A.u[v2 + 4] = *(const unsigned int*)(ap + 16 + 2 * v2);
    }
    const unsigned short* b0 = lA + kc + hiA;        // two 16B runs in LDS
    const unsigned short* b1 = lB + kc + hiA;
#pragma unroll
    for (int v2 = 0; v2 < 4; ++v2) {
      B0.u[v2]     = *(const unsigned int*)(b0 + 2 * v2);
      B0.u[v2 + 4] = *(const unsigned int*)(b0 + 16 + 2 * v2);
      B1.u[v2]     = *(const unsigned int*)(b1 + 2 * v2);
      B1.u[v2 + 4] = *(const unsigned int*)(b1 + 16 + 2 * v2);
    }
    acc0 = __builtin_amdgcn_wmma_f32_16x16x32_bf16(false, A.v, false, B0.v,
                                                   (short)0, acc0, false, false);
    acc1 = __builtin_amdgcn_wmma_f32_16x16x32_bf16(false, A.v, false, B1.v,
                                                   (short)0, acc1, false, false);
  }

  const float biasA = bd[nA], biasB = bd[nB];
  const int rbase = mt * 16 + hiA;
  const size_t tOff = (size_t)t * V_;
#pragma unroll
  for (int r = 0; r < 8; ++r) {
    size_t ro = (size_t)(rbase + r) * ((size_t)T_ * V_) + tOff;
    out[ro + nA] = acc0[r] + biasA;
    out[ro + nB] = acc1[r] + biasB;
  }
}

// ---------------------------------------------------------------------------
// Greedy argmax over V per batch row; first-index tie-break like jnp.argmax.
// ---------------------------------------------------------------------------
__global__ void k_argmax(const float* __restrict__ out, int t,
                         int* __restrict__ idx) {
  __shared__ float sv[256];
  __shared__ int   si[256];
  const int b = blockIdx.x;
  const float* row = out + (size_t)b * ((size_t)T_ * V_) + (size_t)t * V_;
  float best = -3.402823466e38f;
  int bi = 0;
  for (int v = threadIdx.x; v < V_; v += 256) {
    float x = row[v];
    if (x > best) { best = x; bi = v; }
  }
  sv[threadIdx.x] = best; si[threadIdx.x] = bi;
  __syncthreads();
  for (int s = 128; s > 0; s >>= 1) {
    if ((int)threadIdx.x < s) {
      float ov = sv[threadIdx.x + s]; int oi = si[threadIdx.x + s];
      if (ov > sv[threadIdx.x] ||
          (ov == sv[threadIdx.x] && oi < si[threadIdx.x])) {
        sv[threadIdx.x] = ov; si[threadIdx.x] = oi;
      }
    }
    __syncthreads();
  }
  if (threadIdx.x == 0) idx[b] = si[0];
}

// ---------------------------------------------------------------------------
extern "C" void kernel_launch(void* const* d_in, const int* in_sizes, int n_in,
                              void* d_out, int out_size, void* d_ws,
                              size_t ws_size, hipStream_t stream) {
  (void)in_sizes; (void)n_in; (void)out_size; (void)ws_size;
  const float* h0  = (const float*)d_in[0];
  const float* c0  = (const float*)d_in[1];
  const float* emb = (const float*)d_in[2];
  const float* Wx  = (const float*)d_in[3];
  const float* Wh  = (const float*)d_in[4];
  const float* bl  = (const float*)d_in[5];
  const float* Wd  = (const float*)d_in[6];
  const float* bd  = (const float*)d_in[7];
  float* out = (float*)d_out;

  // Workspace carve-up (~37.8 MB). WdT (32.8 MB bf16) fits in 192 MB L2.
  char* w = (char*)d_ws;
  size_t off = 0;
  unsigned short* WdT = (unsigned short*)(w + off); off += (size_t)U_ * V_ * 2;
  unsigned short* WxT = (unsigned short*)(w + off); off += (size_t)E_ * G_ * 2;
  unsigned short* WhT = (unsigned short*)(w + off); off += (size_t)U_ * G_ * 2;
  unsigned short* hbf = (unsigned short*)(w + off); off += (size_t)B_ * U_ * 2;
  unsigned short* xbf = (unsigned short*)(w + off); off += (size_t)B_ * E_ * 2;
  float* cst = (float*)(w + off);                   off += (size_t)B_ * U_ * 4;
  float* z   = (float*)(w + off);                   off += (size_t)B_ * G_ * 4;
  int*   idx = (int*)(w + off);                     off += 256;

  // One-time (per call) weight conversion to bf16, K-major per column.
  k_convert_transpose<<<(V_ * U_ + 255) / 256, 256, 0, stream>>>(Wd, WdT, V_, V_ * U_);
  k_convert_transpose<<<(G_ * E_ + 255) / 256, 256, 0, stream>>>(Wx, WxT, G_, G_ * E_);
  k_convert_transpose<<<(G_ * U_ + 255) / 256, 256, 0, stream>>>(Wh, WhT, G_, G_ * U_);
  k_init_state<<<(B_ * U_) / 256, 256, 0, stream>>>(h0, c0, hbf, cst, idx);

  // Sequential greedy decode: stream order gives the step-to-step dependency.
  for (int t = 0; t < T_; ++t) {
    k_embed<<<(B_ * E_) / 256, 256, 0, stream>>>(emb, idx, xbf);
    k_gates_gemm<<<64, 256, 0, stream>>>(xbf, WxT, WhT, hbf, bl, z);
    k_lstm_update<<<(B_ * U_) / 256, 256, 0, stream>>>(z, cst, hbf);
    k_logits_gemm<<<V_ / 64, 256, 0, stream>>>(hbf, WdT, bd, out, t);
    k_argmax<<<B_, 256, 0, stream>>>(out, t, idx);
  }
}
